// MOE_32384053412169
// MI455X (gfx1250) — matrix operand order repaired and checked by reading, not compiled, not run on hardware
//
#include <hip/hip_runtime.h>

#define N_NODES 32768
#define D_DIM   512
#define N_EDGES 262144

typedef __attribute__((ext_vector_type(16))) __bf16 v16bf;
typedef __attribute__((ext_vector_type(8)))  float  v8f;

__device__ __forceinline__ unsigned short f2bf(float f) {
    unsigned int u = __float_as_uint(f);
    u += 0x7FFFu + ((u >> 16) & 1u);      // round-to-nearest-even
    return (unsigned short)(u >> 16);
}
__device__ __forceinline__ float bf2f(unsigned short h) {
    return __uint_as_float((unsigned int)h << 16);
}

// ---------------------------------------------------------------- utilities
__global__ void zero_kernel(float* __restrict__ p, int n) {
    for (int i = blockIdx.x * blockDim.x + threadIdx.x; i < n;
         i += gridDim.x * blockDim.x)
        p[i] = 0.0f;
}

__global__ void deg_kernel(const int* __restrict__ ei, float* __restrict__ deg) {
    int i = blockIdx.x * 256 + threadIdx.x;
    if (i < N_EDGES) atomicAdd(&deg[ei[N_EDGES + i]], 1.0f);
}

__global__ void invdeg_kernel(float* __restrict__ deg) {
    int i = blockIdx.x * 256 + threadIdx.x;
    if (i < N_NODES) {
        float d = deg[i];
        deg[i] = (d > 0.0f) ? (1.0f / d) : 0.0f;
    }
}

// fp32 -> bf16 bulk convert
__global__ void f2bf_kernel(const float* __restrict__ in,
                            unsigned short* __restrict__ out, int n) {
    int i = blockIdx.x * 256 + threadIdx.x;
    if (i < n) out[i] = f2bf(in[i]);
}

// abf[n][k] = bf16(aggr[n][k] * invdeg[n])
__global__ void aggr2bf_kernel(const float* __restrict__ aggr,
                               const float* __restrict__ invdeg,
                               unsigned short* __restrict__ out) {
    int i = blockIdx.x * 256 + threadIdx.x;      // grid = ND/256
    out[i] = f2bf(aggr[i] * invdeg[i >> 9]);
}

// transpose + convert all 16 weight matrices: wt[m][n][k] = bf16(W[m][k][n])
__global__ void wt_kernel(const float* __restrict__ Wl,
                          const float* __restrict__ Wr,
                          unsigned short* __restrict__ wt) {
    int m = blockIdx.y;                           // 0..15
    const float* src = (m < 8) ? (Wl + (size_t)m * 262144)
                               : (Wr + (size_t)(m - 8) * 262144);
    unsigned short* dst = wt + (size_t)m * 262144;
    int i = blockIdx.x * 256 + threadIdx.x;       // grid.x = 1024
    int n = i >> 9, k = i & 511;
    dst[n * 512 + k] = f2bf(src[k * 512 + n]);
}

// ---------------------------------------------------------------- gating
__global__ __launch_bounds__(256) void gate_kernel(
    const float* __restrict__ x, const float* __restrict__ gW,
    const float* __restrict__ gb, float* __restrict__ coefs)
{
    int wave = threadIdx.x >> 5;
    int lane = threadIdx.x & 31;
    int node = blockIdx.x * 8 + wave;

    float a0 = 0.f, a1 = 0.f, a2 = 0.f, a3 = 0.f;
    const float* xr = x + (size_t)node * D_DIM + lane * 16;
    #pragma unroll
    for (int i = 0; i < 16; ++i) {
        float xv = xr[i];
        const float4 w = *(const float4*)&gW[(lane * 16 + i) * 4];
        a0 += xv * w.x; a1 += xv * w.y; a2 += xv * w.z; a3 += xv * w.w;
    }
    #pragma unroll
    for (int off = 16; off >= 1; off >>= 1) {
        a0 += __shfl_xor(a0, off, 32);
        a1 += __shfl_xor(a1, off, 32);
        a2 += __shfl_xor(a2, off, 32);
        a3 += __shfl_xor(a3, off, 32);
    }
    if (lane == 0) {
        float lg[4] = {a0 + gb[0], a1 + gb[1], a2 + gb[2], a3 + gb[3]};
        float m = fmaxf(fmaxf(lg[0], lg[1]), fmaxf(lg[2], lg[3]));
        float p[4], s = 0.f;
        #pragma unroll
        for (int e = 0; e < 4; ++e) { p[e] = __expf(lg[e] - m); s += p[e]; }
        float inv = 1.0f / s;
        #pragma unroll
        for (int e = 0; e < 4; ++e) p[e] *= inv;
        int i0 = 0;
        for (int e = 1; e < 4; ++e) if (p[e] > p[i0]) i0 = e;   // first max
        int i1 = -1;
        for (int e = 0; e < 4; ++e) {
            if (e == i0) continue;
            if (i1 < 0 || p[e] > p[i1]) i1 = e;
        }
        float c[4] = {0.f, 0.f, 0.f, 0.f};
        c[i0] = p[i0]; c[i1] = p[i1];
        #pragma unroll
        for (int e = 0; e < 4; ++e) coefs[e * N_NODES + node] = c[e];
    }
}

// ---------------------------------------------------------------- scatter
// gather h (bf16) along src, atomic-add fp32 into aggr[dst]
__global__ __launch_bounds__(256) void scatter_add_bf(
    const unsigned short* __restrict__ h, const int* __restrict__ ei,
    float* __restrict__ aggr)
{
    int t = threadIdx.x;
    int edge = blockIdx.x * 2 + (t >> 7);
    int j = (t & 127) * 4;
    int src = ei[edge];
    int dst = ei[N_EDGES + edge];
    const ushort4 v = *(const ushort4*)&h[(size_t)src * D_DIM + j];
    float* a = &aggr[(size_t)dst * D_DIM + j];
    atomicAdd(a + 0, bf2f(v.x)); atomicAdd(a + 1, bf2f(v.y));
    atomicAdd(a + 2, bf2f(v.z)); atomicAdd(a + 3, bf2f(v.w));
}

// ---------------------------------------------------------------- fused GEMM
// C = relu( A1 @ WlT' + A2 @ WrT' + bias )   (A in bf16, weights pre-transposed)
// mode 0: outh = bf16(C)                      (layer-0 output)
// mode 1: outf += coef[row] * C               (layer-1 + MoE combine)
// Tile 128(M) x 64(N), K-step 32, 8 waves; async global->LDS double-buffered.
//
// LDS layout per buffer (halves): A1[128][32] @0, A2[128][32] @4096,
//                                 B1[64][32] @8192, B2[64][32] @10240
#define BUF_H 12288   // halves per buffer (24KB)

__device__ __forceinline__ void async_b128(unsigned lds, int gbyte_off,
                                           const unsigned short* base) {
    asm volatile("global_load_async_to_lds_b128 %0, %1, %2"
                 :: "v"(lds), "v"(gbyte_off), "s"(base) : "memory");
}

__device__ __forceinline__ v16bf ldA(const unsigned short* s, int off) {
    union { v16bf v; uint4 q[2]; } u;
    u.q[0] = *(const uint4*)(s + off);
    u.q[1] = *(const uint4*)(s + off + 8);
    return u.v;
}

__global__ __launch_bounds__(256) void fused_sage_gemm(
    const unsigned short* __restrict__ a1bf,  // aggregated (scaled) feats, bf16
    const unsigned short* __restrict__ a2bf,  // root feats, bf16
    const unsigned short* __restrict__ wtl,   // Wl^T bf16 [n][k]
    const unsigned short* __restrict__ wtr,   // Wr^T bf16 [n][k]
    const float* __restrict__ bias,
    const float* __restrict__ coef,
    float* __restrict__ outf, unsigned short* __restrict__ outh, int mode)
{
    __shared__ __align__(32) unsigned short smem[2 * BUF_H];

    const int tid  = threadIdx.x;
    const int wave = tid >> 5;
    const int lane = tid & 31;
    const int rowBase = blockIdx.x * 128;
    const int colBase = blockIdx.y * 64;

    // per-thread async-copy coordinates (16B chunks)
    const int arow0 = tid >> 2, aseg0 = tid & 3;            // A chunks 0..255
    const int arow1 = (tid + 256) >> 2, aseg1 = tid & 3;    // A chunks 256..511
    const int bn = tid >> 2, bseg = tid & 3;                // B chunks 0..255

    v8f acc[4] = {v8f{}, v8f{}, v8f{}, v8f{}};

    // prologue: stage k-step 0 into buffer 0
    {
        const unsigned sb = (unsigned)(size_t)&smem[0];
        async_b128(sb + (arow0 * 32 + aseg0 * 8) * 2,
                   (((rowBase + arow0) << 9) + aseg0 * 8) * 2, a1bf);
        async_b128(sb + (arow1 * 32 + aseg1 * 8) * 2,
                   (((rowBase + arow1) << 9) + aseg1 * 8) * 2, a1bf);
        async_b128(sb + (4096 + arow0 * 32 + aseg0 * 8) * 2,
                   (((rowBase + arow0) << 9) + aseg0 * 8) * 2, a2bf);
        async_b128(sb + (4096 + arow1 * 32 + aseg1 * 8) * 2,
                   (((rowBase + arow1) << 9) + aseg1 * 8) * 2, a2bf);
        async_b128(sb + (8192 + bn * 32 + bseg * 8) * 2,
                   (((colBase + bn) << 9) + bseg * 8) * 2, wtl);
        async_b128(sb + (10240 + bn * 32 + bseg * 8) * 2,
                   (((colBase + bn) << 9) + bseg * 8) * 2, wtr);
    }

    for (int it = 0; it < 16; ++it) {
        const int cur = it & 1;
        asm volatile("s_wait_asynccnt 0x0" ::: "memory");
        __syncthreads();

        if (it < 15) {  // prefetch next k-step into the other buffer
            const int k1 = (it + 1) * 32;
            const unsigned sb = (unsigned)(size_t)&smem[(1 - cur) * BUF_H];
            async_b128(sb + (arow0 * 32 + aseg0 * 8) * 2,
                       (((rowBase + arow0) << 9) + k1 + aseg0 * 8) * 2, a1bf);
            async_b128(sb + (arow1 * 32 + aseg1 * 8) * 2,
                       (((rowBase + arow1) << 9) + k1 + aseg1 * 8) * 2, a1bf);
            async_b128(sb + (4096 + arow0 * 32 + aseg0 * 8) * 2,
                       (((rowBase + arow0) << 9) + k1 + aseg0 * 8) * 2, a2bf);
            async_b128(sb + (4096 + arow1 * 32 + aseg1 * 8) * 2,
                       (((rowBase + arow1) << 9) + k1 + aseg1 * 8) * 2, a2bf);
            async_b128(sb + (8192 + bn * 32 + bseg * 8) * 2,
                       (((colBase + bn) << 9) + k1 + bseg * 8) * 2, wtl);
            async_b128(sb + (10240 + bn * 32 + bseg * 8) * 2,
                       (((colBase + bn) << 9) + k1 + bseg * 8) * 2, wtr);
        }

        // consume current buffer
        const unsigned short* sb = &smem[cur * BUF_H];
        const int arow = wave * 16 + (lane & 15);
        const int akoff = (lane >= 16) ? 16 : 0;
        v16bf a1 = ldA(sb, arow * 32 + akoff);
        v16bf a2 = ldA(sb + 4096, arow * 32 + akoff);
        const int bk = (lane >= 16) ? 16 : 0;
        #pragma unroll
        for (int t = 0; t < 4; ++t) {
            const int brow = (t * 16 + (lane & 15)) * 32 + bk;
            v16bf b1 = *(const v16bf*)&sb[8192 + brow];
            acc[t] = __builtin_amdgcn_wmma_f32_16x16x32_bf16(
                false, a1, false, b1, (short)0, acc[t], false, false);
            v16bf b2 = *(const v16bf*)&sb[10240 + brow];
            acc[t] = __builtin_amdgcn_wmma_f32_16x16x32_bf16(
                false, a2, false, b2, (short)0, acc[t], false, false);
        }
    }

    // ---- epilogue: +bias, ReLU, store bf16 or gate-weighted f32 accumulate
    const int mrow = rowBase + wave * 16 + ((lane >= 16) ? 8 : 0);
    #pragma unroll
    for (int t = 0; t < 4; ++t) {
        int col = colBase + t * 16 + (lane & 15);
        float bv = bias[col];
        #pragma unroll
        for (int i = 0; i < 8; ++i) {
            int row = mrow + i;
            float v = acc[t][i] + bv;
            v = (v > 0.0f) ? v : 0.0f;
            size_t idx = (size_t)row * D_DIM + col;
            if (mode == 0) outh[idx] = f2bf(v);
            else           outf[idx] += coef[row] * v;
        }
    }
}

// ---------------------------------------------------------------- launcher
extern "C" void kernel_launch(void* const* d_in, const int* in_sizes, int n_in,
                              void* d_out, int out_size, void* d_ws, size_t ws_size,
                              hipStream_t stream)
{
    const float* x   = (const float*)d_in[0];
    const float* gW  = (const float*)d_in[1];
    const float* gb  = (const float*)d_in[2];
    const float* Wl  = (const float*)d_in[3];
    const float* bl  = (const float*)d_in[4];
    const float* Wr  = (const float*)d_in[5];
    const int*   ei  = (const int*)d_in[6];
    float* out = (float*)d_out;

    const int N = N_NODES, Dd = D_DIM, E = N_EDGES;
    const int ND = N * Dd;

    float* ws     = (float*)d_ws;
    float* invdeg = ws;                         // N floats
    float* coefs  = ws + N;                     // 4N floats
    float* aggr   = ws + 5 * N;                 // ND floats (reused L0/L1)
    unsigned short* bfb  = (unsigned short*)(ws + 5 * (size_t)N + ND);
    unsigned short* xbf  = bfb;                 // ND halves
    unsigned short* hbf  = bfb + (size_t)ND;    // ND halves
    unsigned short* abf0 = bfb + 2 * (size_t)ND;
    unsigned short* abf1 = bfb + 3 * (size_t)ND;
    unsigned short* wt   = bfb + 4 * (size_t)ND; // 16 * 512*512 halves

    // init + gating + degrees
    zero_kernel<<<dim3((ND + 255) / 256), 256, 0, stream>>>(out, ND);
    zero_kernel<<<dim3((N + 255) / 256), 256, 0, stream>>>(invdeg, N);
    zero_kernel<<<dim3((ND + 255) / 256), 256, 0, stream>>>(aggr, ND);
    gate_kernel<<<dim3(N / 8), 256, 0, stream>>>(x, gW, gb, coefs);
    deg_kernel<<<dim3(E / 256), 256, 0, stream>>>(ei, invdeg);
    invdeg_kernel<<<dim3(N / 256), 256, 0, stream>>>(invdeg);

    // one-time bf16 conversions: x and all 16 transposed weight matrices
    f2bf_kernel<<<dim3(ND / 256), 256, 0, stream>>>(x, xbf, ND);
    wt_kernel<<<dim3(1024, 16), 256, 0, stream>>>(Wl, Wr, wt);

    // layer-0 aggregation (expert independent): aggr = sum x[src] -> abf0
    scatter_add_bf<<<dim3(E / 2), 256, 0, stream>>>(xbf, ei, aggr);
    aggr2bf_kernel<<<dim3(ND / 256), 256, 0, stream>>>(aggr, invdeg, abf0);

    dim3 gemm_grid(N / 128, Dd / 64);
    for (int e = 0; e < 4; ++e) {
        const unsigned short* wtl0 = wt + (size_t)(e * 2 + 0) * 262144;
        const unsigned short* wtl1 = wt + (size_t)(e * 2 + 1) * 262144;
        const unsigned short* wtr0 = wt + (size_t)(8 + e * 2 + 0) * 262144;
        const unsigned short* wtr1 = wt + (size_t)(8 + e * 2 + 1) * 262144;
        const float* b0 = bl + (size_t)(e * 2 + 0) * Dd;
        const float* b1 = bl + (size_t)(e * 2 + 1) * Dd;

        // layer 0: hbf = bf16(relu(mean_agg(x)@Wl0 + x@Wr0 + b0))
        fused_sage_gemm<<<gemm_grid, 256, 0, stream>>>(
            abf0, xbf, wtl0, wtr0, b0, coefs, out, hbf, 0);

        // layer 1 aggregation of h
        zero_kernel<<<dim3((ND + 255) / 256), 256, 0, stream>>>(aggr, ND);
        scatter_add_bf<<<dim3(E / 2), 256, 0, stream>>>(hbf, ei, aggr);
        aggr2bf_kernel<<<dim3(ND / 256), 256, 0, stream>>>(aggr, invdeg, abf1);

        // layer 1 + MoE combine
        fused_sage_gemm<<<gemm_grid, 256, 0, stream>>>(
            abf1, hbf, wtl1, wtr1, b1, coefs + e * N, out, hbf, 1);
    }
}
